// AttentionBlock_2405181686215
// MI455X (gfx1250) — matrix operand order repaired and checked by reading, not compiled, not run on hardware
//
#include <hip/hip_runtime.h>

// ---------- problem constants ----------
#define BB 16
#define CC 512
#define LL 1024
#define NH 8
#define CH 64
#define GG 32
#define OO (3*CC)          // 1536
#define SCALE 0.35355339059327373f  // 1/sqrt(sqrt(64)) applied to q AND k

typedef _Float16 h2   __attribute__((ext_vector_type(2)));
typedef _Float16 v16h __attribute__((ext_vector_type(16)));
typedef float    v8f  __attribute__((ext_vector_type(8)));

__device__ __forceinline__ v8f wmma16x16x32(v16h a, v16h b, v8f c) {
  return __builtin_amdgcn_wmma_f32_16x16x32_f16(false, a, false, b, (short)0, c, false, false);
}

// A fragment (16x32 f16): lane row m = lane&15; hl = lane>>4.
// VGPR word j holds K = hl*16 + 2j , +1 (contiguous, matches B)
__device__ __forceinline__ v16h afrag_row(const _Float16* prow, int hl) {
  v16h f;
#pragma unroll
  for (int j = 0; j < 8; ++j) {
    int k = hl * 16 + 2 * j;
    h2 v = *(const h2*)(prow + k);
    f[2 * j] = v[0]; f[2 * j + 1] = v[1];
  }
  return f;
}

// B fragment (32x16 f16): lane col n = lane&15; hl = lane>>4.
// word j holds K = hl*16 + 2j , +1. Caller folds hl*16 into pc.
__device__ __forceinline__ v16h bfrag_contig(const _Float16* pc) {
  v16h f;
#pragma unroll
  for (int j = 0; j < 8; ++j) {
    h2 v = *(const h2*)(pc + 2 * j);
    f[2 * j] = v[0]; f[2 * j + 1] = v[1];
  }
  return f;
}

// gfx1250 async global->LDS copy, 16 bytes, tracked by ASYNCcnt.
__device__ __forceinline__ void async_copy16(void* lds_ptr, const void* gptr) {
  unsigned lds = (unsigned)(uintptr_t)lds_ptr;       // low 32 bits = LDS offset
  unsigned long long ga = (unsigned long long)(uintptr_t)gptr;
  asm volatile("global_load_async_to_lds_b128 %0, %1, off"
               :: "v"(lds), "v"(ga) : "memory");
}
__device__ __forceinline__ void wait_async0() {
  asm volatile("s_wait_asynccnt 0" ::: "memory");
}

// ---------------- K1: GroupNorm -> xn_t [b][l][c] f16 ----------------
__global__ void gn_kernel(const float* __restrict__ x,
                          const float* __restrict__ nw,
                          const float* __restrict__ nb,
                          _Float16* __restrict__ xn_t) {
  int gid = blockIdx.x;            // b*32 + g
  int b = gid >> 5, g = gid & 31;
  const float* xb = x + ((size_t)b * CC + (size_t)g * 16) * LL;  // 16 ch x 1024, contiguous
  float s = 0.f, ss = 0.f;
  for (int i = threadIdx.x; i < 16 * LL; i += 256) {
    float v = xb[i];
    s += v; ss += v * v;
  }
#pragma unroll
  for (int m = 16; m >= 1; m >>= 1) {
    s  += __shfl_xor(s,  m, 32);
    ss += __shfl_xor(ss, m, 32);
  }
  __shared__ float rs[8], rss[8];
  int wv = threadIdx.x >> 5;
  if ((threadIdx.x & 31) == 0) { rs[wv] = s; rss[wv] = ss; }
  __syncthreads();
  if (threadIdx.x == 0) {
    float a = 0.f, q = 0.f;
#pragma unroll
    for (int i = 0; i < 8; ++i) { a += rs[i]; q += rss[i]; }
    rs[0] = a; rss[0] = q;
  }
  __syncthreads();
  float mean = rs[0] * (1.f / 16384.f);
  float var  = rss[0] * (1.f / 16384.f) - mean * mean;
  float rstd = rsqrtf(var + 1e-5f);
  for (int i = threadIdx.x; i < 16 * LL; i += 256) {
    int cc = i >> 10, l = i & (LL - 1);
    int c = g * 16 + cc;
    float v = (xb[i] - mean) * rstd * nw[c] + nb[c];
    xn_t[((size_t)b * LL + l) * CC + c] = (_Float16)v;
  }
}

// ---------------- K2: weight conversion fp32 -> f16 ----------------
__global__ void cvt_w_kernel(const float* __restrict__ qkvw,
                             const float* __restrict__ projw,
                             _Float16* __restrict__ qkvwh,
                             _Float16* __restrict__ projwh) {
  int i = blockIdx.x * 256 + threadIdx.x;
  if (i < OO * CC)  qkvwh[i]  = (_Float16)qkvw[i];
  if (i < CC * CC)  projwh[i] = (_Float16)projw[i];
}

// ---------------- K3: QKV GEMM, 2x2 register-blocked (32x32 per wave) ----
// W[1536x512] x Xn[512x1024] per batch. Scatter q->[b][h][t][c],
// k->[b][h][s][c], v->[b][h][c][s]; q,k pre-scaled by SCALE.
__global__ void qkv_gemm_kernel(const _Float16* __restrict__ wh,
                                const float* __restrict__ bias,
                                const _Float16* __restrict__ xn_t,
                                _Float16* __restrict__ q_t,
                                _Float16* __restrict__ k_t,
                                _Float16* __restrict__ v_t) {
  int wid  = blockIdx.x * 8 + (threadIdx.x >> 5);
  int lane = threadIdx.x & 31;
  int nt = wid & 31; int tmp = wid >> 5;          // 32 n-tiles of 32
  int mt = tmp % 48; int b = tmp / 48;            // 48 m-tiles of 32
  int m0 = mt * 32, n0 = nt * 32;
  int mrow = lane & 15, hl = lane >> 4;

  const _Float16* arow0 = wh + (size_t)(m0 + mrow) * CC;
  const _Float16* arow1 = arow0 + (size_t)16 * CC;
  const _Float16* brow0 = xn_t + ((size_t)b * LL + n0 + mrow) * CC + hl * 16;
  const _Float16* brow1 = brow0 + (size_t)16 * CC;

  v8f acc[2][2] = {};
  for (int k0 = 0; k0 < CC; k0 += 32) {
    __builtin_prefetch(arow0 + k0 + 128, 0, 3);
    __builtin_prefetch(brow0 + k0 + 128, 0, 3);
    v16h af0 = afrag_row(arow0 + k0, hl);
    v16h af1 = afrag_row(arow1 + k0, hl);
    v16h bf0 = bfrag_contig(brow0 + k0);
    v16h bf1 = bfrag_contig(brow1 + k0);
    acc[0][0] = wmma16x16x32(af0, bf0, acc[0][0]);
    acc[0][1] = wmma16x16x32(af0, bf1, acc[0][1]);
    acc[1][0] = wmma16x16x32(af1, bf0, acc[1][0]);
    acc[1][1] = wmma16x16x32(af1, bf1, acc[1][1]);
  }
#pragma unroll
  for (int mi = 0; mi < 2; ++mi) {
#pragma unroll
    for (int ni = 0; ni < 2; ++ni) {
      int l = n0 + ni * 16 + mrow;
#pragma unroll
      for (int r = 0; r < 8; ++r) {
        int o = m0 + mi * 16 + r + hl * 8;
        float v = acc[mi][ni][r] + bias[o];
        if (o < CC) {
          int h = o >> 6, c = o & 63;
          q_t[(((size_t)b * NH + h) * LL + l) * CH + c] = (_Float16)(v * SCALE);
        } else if (o < 2 * CC) {
          int o2 = o - CC; int h = o2 >> 6, c = o2 & 63;
          k_t[(((size_t)b * NH + h) * LL + l) * CH + c] = (_Float16)(v * SCALE);
        } else {
          int o2 = o - 2 * CC; int h = o2 >> 6, c = o2 & 63;
          v_t[(((size_t)b * NH + h) * CH + c) * LL + l] = (_Float16)v;
        }
      }
    }
  }
}

// ---------------- K4: flash attention with async-LDS K/V staging --------
// WG = 8 waves; all share one (b,h). Each wave owns 16 query rows.
// Per 32-key tile: 8 KB staged to LDS via global_load_async_to_lds_b128.
__global__ void attn_kernel(const _Float16* __restrict__ q_t,
                            const _Float16* __restrict__ k_t,
                            const _Float16* __restrict__ v_t,
                            _Float16* __restrict__ a_t) {
  __shared__ _Float16 Kt[32][72];   // [s_local][c], 144B rows: 16B-aligned, conflict-free
  __shared__ _Float16 Vt[64][40];   // [c][s_local], 80B rows
  __shared__ _Float16 P[8][16][40]; // per-wave P tile
  int wv = threadIdx.x >> 5, lane = threadIdx.x & 31;
  int tid = threadIdx.x;
  int bid = blockIdx.x;
  int tpart = bid & 7; int h = (bid >> 3) & 7; int b = bid >> 6;
  int t0 = tpart * 128 + wv * 16;
  int mrow = lane & 15, hl = lane >> 4;
  int n = mrow;

  const _Float16* qbase = q_t + (((size_t)b * NH + h) * LL) * CH;
  const _Float16* kbase = k_t + (((size_t)b * NH + h) * LL) * CH;
  const _Float16* vbase = v_t + (((size_t)b * NH + h) * CH) * LL;

  // async copy assignments: K = 32 rows x 8 chunks of 16B; V = 64 rows x 4 chunks
  int krow = tid >> 3, kchk = tid & 7;
  int vrow = tid >> 2, vchk = tid & 3;

  // Q A-fragments (rows t0..t0+15, c split 0..31 / 32..63)
  v16h qa0, qa1;
  {
    const _Float16* qrow = qbase + (size_t)(t0 + mrow) * CH;
    qa0 = afrag_row(qrow, hl);
    qa1 = afrag_row(qrow + 32, hl);
  }

  float mrun[8], lrun[8];
#pragma unroll
  for (int r = 0; r < 8; ++r) { mrun[r] = -1e30f; lrun[r] = 0.f; }
  v8f o0 = {}, o1 = {}, o2 = {}, o3 = {};

  for (int s0 = 0; s0 < LL; s0 += 32) {
    __syncthreads();   // previous tile fully consumed by all waves
    async_copy16(&Kt[krow][kchk * 8], kbase + (size_t)(s0 + krow) * CH + kchk * 8);
    async_copy16(&Vt[vrow][vchk * 8], vbase + (size_t)vrow * LL + s0 + vchk * 8);
    wait_async0();
    __syncthreads();   // all waves' copies landed

    // S = Q K^T for 32 keys (fragments read from LDS)
    v8f S0 = {}, S1 = {};
    {
      const _Float16* c0 = &Kt[n][hl * 16];
      const _Float16* c1 = &Kt[n + 16][hl * 16];
      S0 = wmma16x16x32(qa0, bfrag_contig(c0), S0);
      S0 = wmma16x16x32(qa1, bfrag_contig(c0 + 32), S0);
      S1 = wmma16x16x32(qa0, bfrag_contig(c1), S1);
      S1 = wmma16x16x32(qa1, bfrag_contig(c1 + 32), S1);
    }
    // online softmax (rows live in half-specific lanes: row = r + hl*8)
    float al[8];
#pragma unroll
    for (int r = 0; r < 8; ++r) {
      float v = fmaxf(S0[r], S1[r]);
#pragma unroll
      for (int msk = 1; msk < 16; msk <<= 1) v = fmaxf(v, __shfl_xor(v, msk, 32));
      float mnew = fmaxf(mrun[r], v);
      float alpha = __expf(mrun[r] - mnew);
      float p0 = __expf(S0[r] - mnew);
      float p1 = __expf(S1[r] - mnew);
      float rsum = p0 + p1;
#pragma unroll
      for (int msk = 1; msk < 16; msk <<= 1) rsum += __shfl_xor(rsum, msk, 32);
      lrun[r] = lrun[r] * alpha + rsum;
      mrun[r] = mnew;
      al[r] = alpha;
      S0[r] = p0; S1[r] = p1;
    }
#pragma unroll
    for (int r = 0; r < 8; ++r) {
      o0[r] *= al[r]; o1[r] *= al[r]; o2[r] *= al[r]; o3[r] *= al[r];
    }
    // transpose P (C-layout) -> A-layout via per-wave LDS tile
#pragma unroll
    for (int r = 0; r < 8; ++r) {
      P[wv][r + hl * 8][n]      = (_Float16)S0[r];
      P[wv][r + hl * 8][n + 16] = (_Float16)S1[r];
    }
    asm volatile("s_wait_dscnt 0" ::: "memory");   // LDS in-order per wave
    v16h pa = afrag_row(&P[wv][mrow][0], hl);
    // O += P * V^T (K=s, 32 keys; N = 64 channels in 4 tiles)
    o0 = wmma16x16x32(pa, bfrag_contig(&Vt[n][hl * 16]),      o0);
    o1 = wmma16x16x32(pa, bfrag_contig(&Vt[16 + n][hl * 16]), o1);
    o2 = wmma16x16x32(pa, bfrag_contig(&Vt[32 + n][hl * 16]), o2);
    o3 = wmma16x16x32(pa, bfrag_contig(&Vt[48 + n][hl * 16]), o3);
  }

  // normalize rows and store a[b][t][h*64 + c] (f16) for the proj GEMM
#pragma unroll
  for (int r = 0; r < 8; ++r) {
    float inv = 1.0f / lrun[r];
    int t = t0 + r + hl * 8;
    _Float16* ap = a_t + ((size_t)b * LL + t) * CC + h * CH + n;
    ap[0]  = (_Float16)(o0[r] * inv);
    ap[16] = (_Float16)(o1[r] * inv);
    ap[32] = (_Float16)(o2[r] * inv);
    ap[48] = (_Float16)(o3[r] * inv);
  }
}

// ---------------- K5: proj GEMM, 2x2 blocked + bias + residual ----------
__global__ void proj_gemm_kernel(const _Float16* __restrict__ pwh,
                                 const float* __restrict__ bias,
                                 const _Float16* __restrict__ a_t,
                                 const float* __restrict__ x,
                                 float* __restrict__ out) {
  int wid  = blockIdx.x * 8 + (threadIdx.x >> 5);
  int lane = threadIdx.x & 31;
  int nt = wid & 31; int tmp = wid >> 5;          // 32 n-tiles of 32
  int mt = tmp & 15; int b = tmp >> 4;            // 16 m-tiles of 32
  int m0 = mt * 32, n0 = nt * 32;
  int mrow = lane & 15, hl = lane >> 4;

  const _Float16* arow0 = pwh + (size_t)(m0 + mrow) * CC;
  const _Float16* arow1 = arow0 + (size_t)16 * CC;
  const _Float16* brow0 = a_t + ((size_t)b * LL + n0 + mrow) * CC + hl * 16;
  const _Float16* brow1 = brow0 + (size_t)16 * CC;

  v8f acc[2][2] = {};
  for (int k0 = 0; k0 < CC; k0 += 32) {
    __builtin_prefetch(brow0 + k0 + 128, 0, 3);
    v16h af0 = afrag_row(arow0 + k0, hl);
    v16h af1 = afrag_row(arow1 + k0, hl);
    v16h bf0 = bfrag_contig(brow0 + k0);
    v16h bf1 = bfrag_contig(brow1 + k0);
    acc[0][0] = wmma16x16x32(af0, bf0, acc[0][0]);
    acc[0][1] = wmma16x16x32(af0, bf1, acc[0][1]);
    acc[1][0] = wmma16x16x32(af1, bf0, acc[1][0]);
    acc[1][1] = wmma16x16x32(af1, bf1, acc[1][1]);
  }
#pragma unroll
  for (int mi = 0; mi < 2; ++mi) {
#pragma unroll
    for (int ni = 0; ni < 2; ++ni) {
      int l = n0 + ni * 16 + mrow;
#pragma unroll
      for (int r = 0; r < 8; ++r) {
        int o = m0 + mi * 16 + r + hl * 8;
        size_t idx = ((size_t)b * CC + o) * LL + l;
        out[idx] = x[idx] + acc[mi][ni][r] + bias[o];
      }
    }
  }
}

// ---------------- launch ----------------
extern "C" void kernel_launch(void* const* d_in, const int* in_sizes, int n_in,
                              void* d_out, int out_size, void* d_ws, size_t ws_size,
                              hipStream_t stream) {
  const float* x      = (const float*)d_in[0];
  const float* norm_w = (const float*)d_in[1];
  const float* norm_b = (const float*)d_in[2];
  const float* qkv_w  = (const float*)d_in[3];
  const float* qkv_b  = (const float*)d_in[4];
  const float* proj_w = (const float*)d_in[5];
  const float* proj_b = (const float*)d_in[6];
  float* out = (float*)d_out;

  char* ws = (char*)d_ws;
  const size_t sz_xn = (size_t)BB * LL * CC * sizeof(_Float16);      // 16.78 MB
  const size_t sz_q  = (size_t)BB * NH * LL * CH * sizeof(_Float16); // 16.78 MB
  const size_t sz_wq = (size_t)OO * CC * sizeof(_Float16);
  const size_t sz_wp = (size_t)CC * CC * sizeof(_Float16);
  size_t off = 0;
  _Float16* xn_t = (_Float16*)(ws + off); off += sz_xn;
  _Float16* q_t  = (_Float16*)(ws + off); off += sz_q;
  _Float16* k_t  = (_Float16*)(ws + off); off += sz_q;
  _Float16* v_t  = (_Float16*)(ws + off); off += sz_q;
  _Float16* a_t  = (_Float16*)(ws + off); off += sz_xn;
  _Float16* wqh  = (_Float16*)(ws + off); off += sz_wq;
  _Float16* wph  = (_Float16*)(ws + off); off += sz_wp;
  if (off > ws_size) return;  // workspace too small; nothing safe to do

  // K1: GroupNorm (B*32 groups)
  gn_kernel<<<BB * GG, 256, 0, stream>>>(x, norm_w, norm_b, xn_t);
  // K2: weight conversion
  cvt_w_kernel<<<(OO * CC + 255) / 256, 256, 0, stream>>>(qkv_w, proj_w, wqh, wph);
  // K3: QKV GEMM — waves = B * 48 * 32 = 24576 -> 3072 WGs
  qkv_gemm_kernel<<<(BB * 48 * 32) / 8, 256, 0, stream>>>(wqh, qkv_b, xn_t, q_t, k_t, v_t);
  // K4: attention — B * NH * (L/128) = 1024 WGs
  attn_kernel<<<BB * NH * (LL / 128), 256, 0, stream>>>(q_t, k_t, v_t, a_t);
  // K5: proj GEMM — waves = B * 16 * 32 = 8192 -> 1024 WGs
  proj_gemm_kernel<<<(BB * 16 * 32) / 8, 256, 0, stream>>>(wph, proj_b, a_t, x, out);
}
